// NodeEdgeConv_33406255629214
// MI455X (gfx1250) — matrix operand order repaired and checked by reading, not compiled, not run on hardware
//
#include <hip/hip_runtime.h>

// CDNA5 / gfx1250, wave32. D = A(MxK) * B(KxN) [+ Cadd] [+ bias], bf16 WMMA, f32 accumulate.
// Workgroup tile 128x128, BK=32, 8 waves, each wave owns 32x64 (2x4 WMMA accumulators).

typedef __attribute__((ext_vector_type(16))) __bf16 v16bf;
typedef __attribute__((ext_vector_type(8)))  __bf16 v8bf;
typedef __attribute__((ext_vector_type(8)))  float  v8f;
typedef __attribute__((ext_vector_type(4)))  float  v4f;

#define BM 128
#define BN 128
#define BK 32
#define AS 40    // padded LDS row stride for A (bf16): 80 B/row
#define BS 136   // padded LDS row stride for B (bf16): 272 B/row -> lane->bank step 4

template<bool HASC, bool HASB>
__global__ __launch_bounds__(256)
void gemm_bf16_wmma(const float* __restrict__ A, const float* __restrict__ B,
                    const float* __restrict__ Cadd, const float* __restrict__ bias,
                    float* __restrict__ D,
                    int M, int N, int K,
                    long long sA, long long sB, long long sC, long long sD)
{
    const int tid   = threadIdx.x;
    const int lane  = tid & 31;
    const int wave  = tid >> 5;
    const int wm    = wave & 3;   // 4 waves along M, 32 rows each
    const int wn    = wave >> 2;  // 2 waves along N, 64 cols each
    const int bz    = blockIdx.z;

    const float* Ab = A + (long long)bz * sA;
    const float* Bb = B + (long long)bz * sB;
    const float* Cb = HASC ? (Cadd + (long long)bz * sC) : nullptr;
    float*       Db = D + (long long)bz * sD;

    const int m0 = blockIdx.x * BM;
    const int n0 = blockIdx.y * BN;

    __shared__ __bf16 Asu[BM * AS];   // 10.0 KB
    __shared__ __bf16 Bsu[BK * BS];   // 8.5 KB

    v8f zero = {};
    v8f acc[2][4];
    #pragma unroll
    for (int i = 0; i < 2; ++i)
        #pragma unroll
        for (int j = 0; j < 4; ++j) acc[i][j] = zero;

    const int l15   = lane & 15;
    const int khalf = lane >> 4;

    // -------- register-prefetch pipeline --------
    v4f pa[4], pb[4];

    auto loadTiles = [&](int k0) {
        #pragma unroll
        for (int i = 0; i < 4; ++i) {            // A: 128x32 f32 = 1024 float4
            int idx = tid + i * 256;
            int r = idx >> 3, c = idx & 7;
            pa[i] = *(const v4f*)(Ab + (long long)(m0 + r) * K + k0 + c * 4);
        }
        #pragma unroll
        for (int i = 0; i < 4; ++i) {            // B: 32x128 f32 = 1024 float4
            int idx = tid + i * 256;
            int r = idx >> 5, c = idx & 31;
            pb[i] = *(const v4f*)(Bb + (long long)(k0 + r) * N + n0 + c * 4);
        }
    };
    auto stageTiles = [&]() {
        #pragma unroll
        for (int i = 0; i < 4; ++i) {
            int idx = tid + i * 256;
            int r = idx >> 3, c = idx & 7;
            __bf16* p = &Asu[r * AS + c * 4];
            p[0] = (__bf16)pa[i].x; p[1] = (__bf16)pa[i].y;
            p[2] = (__bf16)pa[i].z; p[3] = (__bf16)pa[i].w;
        }
        #pragma unroll
        for (int i = 0; i < 4; ++i) {
            int idx = tid + i * 256;
            int r = idx >> 5, c = idx & 31;
            __bf16* p = &Bsu[r * BS + c * 4];
            p[0] = (__bf16)pb[i].x; p[1] = (__bf16)pb[i].y;
            p[2] = (__bf16)pb[i].z; p[3] = (__bf16)pb[i].w;
        }
    };

    loadTiles(0);

    for (int k0 = 0; k0 < K; k0 += BK) {
        __syncthreads();                 // previous tile's LDS reads done
        stageTiles();                    // cvt f32->bf16, ds_store
        __syncthreads();
        if (k0 + BK < K) loadTiles(k0 + BK);   // overlap next-tile global loads with WMMAs

        // A fragment: lane -> row M=(lane&15); K elems {kh*8..+7, 16+kh*8..+7}
        v16bf af[2], bfg[4];
        #pragma unroll
        for (int mi = 0; mi < 2; ++mi) {
            int mloc = wm * 32 + mi * 16 + l15;
            v8bf lo = *(const v8bf*)&Asu[mloc * AS + khalf * 8];
            v8bf hi = *(const v8bf*)&Asu[mloc * AS + 16 + khalf * 8];
            #pragma unroll
            for (int e = 0; e < 8; ++e) { af[mi][e] = lo[e]; af[mi][e + 8] = hi[e]; }
        }
        // B fragment: lane -> col N=(lane&15); K elems {kh*8..+7, 16+kh*8..+7}
        #pragma unroll
        for (int ni = 0; ni < 4; ++ni) {
            int nloc = wn * 64 + ni * 16;
            #pragma unroll
            for (int e = 0; e < 8; ++e) { bfg[ni][e] = Bsu[(khalf * 8 + e) * BS + nloc + l15]; bfg[ni][e + 8] = Bsu[(16 + khalf * 8 + e) * BS + nloc + l15]; }
        }

        #pragma unroll
        for (int mi = 0; mi < 2; ++mi)
            #pragma unroll
            for (int ni = 0; ni < 4; ++ni)
                acc[mi][ni] = __builtin_amdgcn_wmma_f32_16x16x32_bf16(
                    false, af[mi], false, bfg[ni],
                    (short)0, acc[mi][ni], false, false);
    }

    // Epilogue: D element j -> M = j + 8*(lane>=16), N = lane&15
    #pragma unroll
    for (int mi = 0; mi < 2; ++mi) {
        #pragma unroll
        for (int ni = 0; ni < 4; ++ni) {
            int mb = m0 + wm * 32 + mi * 16;
            int n  = n0 + wn * 64 + ni * 16 + l15;
            float bv = 0.f;
            if constexpr (HASB) bv = bias[n];
            float cv[8];
            if constexpr (HASC) {
                #pragma unroll
                for (int j = 0; j < 8; ++j)
                    cv[j] = Cb[(long long)(mb + j + 8 * khalf) * N + n];
            }
            #pragma unroll
            for (int j = 0; j < 8; ++j) {
                int m = mb + j + 8 * khalf;
                float v = acc[mi][ni][j];
                if constexpr (HASC) v += cv[j];
                if constexpr (HASB) v += bv;
                Db[(long long)m * N + n] = v;
            }
        }
    }
}

__global__ __launch_bounds__(256)
void copy_f32x4(const float* __restrict__ src, float* __restrict__ dst, long long n4)
{
    long long i      = (long long)blockIdx.x * blockDim.x + threadIdx.x;
    long long stride = (long long)gridDim.x * blockDim.x;
    for (; i < n4; i += stride)
        ((v4f*)dst)[i] = ((const v4f*)src)[i];
}

extern "C" void kernel_launch(void* const* d_in, const int* in_sizes, int n_in,
                              void* d_out, int out_size, void* d_ws, size_t ws_size,
                              hipStream_t stream)
{
    (void)in_sizes; (void)n_in; (void)out_size; (void)ws_size;

    const int Bc = 8, Cc = 4, V = 1024, E = 2048, F = 256;
    const int BCN = Bc * Cc; // 32 batch slices

    const float* H_v    = (const float*)d_in[0];
    const float* H_e    = (const float*)d_in[1];
    // d_in[2] = adj_e (unused in node_layer branch)
    const float* adj_v  = (const float*)d_in[3];
    const float* T      = (const float*)d_in[4];
    const float* weight = (const float*)d_in[5];
    const float* w_sec  = (const float*)d_in[6];
    const float* bias   = (const float*)d_in[7];

    float* out = (float*)d_out;
    float* X = (float*)d_ws;                       // [32,1024,256]
    float* Y = X + (long long)BCN * V * F;         // [32,1024,256]

    dim3 block(256);
    dim3 grid(V / BM, F / BN, BCN);                // (8, 2, 32)
    const long long sVF = (long long)V * F;
    const long long sEF = (long long)E * F;

    // 1) X = T @ H_e            (M=V, N=F, K=E; A shared)
    gemm_bf16_wmma<false, false><<<grid, block, 0, stream>>>(
        T, H_e, nullptr, nullptr, X, V, F, E, 0, sEF, 0, sVF);
    // 2) Y = H_v + X @ w_sec    (M=V, N=F, K=F; B shared)
    gemm_bf16_wmma<true, false><<<grid, block, 0, stream>>>(
        X, w_sec, H_v, nullptr, Y, V, F, F, sVF, 0, sVF, sVF);
    // 3) Z = Y @ weight  (Z reuses X)
    gemm_bf16_wmma<false, false><<<grid, block, 0, stream>>>(
        Y, weight, nullptr, nullptr, X, V, F, F, sVF, 0, 0, sVF);
    // 4) ret = adj_v @ Z + bias (M=V, N=F, K=V; A shared)
    gemm_bf16_wmma<false, true><<<grid, block, 0, stream>>>(
        adj_v, X, nullptr, bias, out, V, F, V, 0, sVF, 0, sVF);
    // 5) second tuple element: H_e passthrough
    long long n4 = (long long)BCN * E * F / 4;
    copy_f32x4<<<1024, block, 0, stream>>>(H_e, out + (long long)BCN * V * F, n4);
}
